// TopologicalCausalHead_89129161326841
// MI455X (gfx1250) — matrix operand order repaired and checked by reading, not compiled, not run on hardware
//
#include <hip/hip_runtime.h>
#include <hip/hip_bf16.h>
#include <math.h>

typedef __attribute__((ext_vector_type(16))) _Float16 v16h;
typedef __attribute__((ext_vector_type(8)))  _Float16 v8h;
typedef __attribute__((ext_vector_type(8)))  float    v8f;

#define D_MODEL 1024
#define SEQ     2048
#define BATCH   4
#define ROWS    (BATCH * SEQ)   // 8192 flattened rows

// ---------------------------------------------------------------------------
// f32 -> f16 bulk convert
// ---------------------------------------------------------------------------
__global__ __launch_bounds__(256) void cvt_f16_kernel(const float* __restrict__ in,
                                                      _Float16* __restrict__ out, int n) {
    int i = blockIdx.x * 256 + threadIdx.x;
    if (i < n) out[i] = (_Float16)in[i];
}

// ---------------------------------------------------------------------------
// LDS-tiled transpose: Wt[n*D + k] = (f16) W[k*D + n]; coalesced both sides.
// block = 256 (32x8), grid = (32, 32) for 1024x1024.
// ---------------------------------------------------------------------------
__global__ __launch_bounds__(256) void transpose_f16_kernel(const float* __restrict__ W,
                                                            _Float16* __restrict__ Wt) {
    __shared__ float tile[32][33];
    const int tx = threadIdx.x & 31;
    const int ty = threadIdx.x >> 5;
    const int bx = blockIdx.x;     // n-tile of Wt
    const int by = blockIdx.y;     // k-tile of Wt
    #pragma unroll
    for (int i = 0; i < 32; i += 8)
        tile[ty + i][tx] = W[(size_t)(by * 32 + ty + i) * D_MODEL + bx * 32 + tx];
    __syncthreads();
    #pragma unroll
    for (int i = 0; i < 32; i += 8)
        Wt[(size_t)(bx * 32 + ty + i) * D_MODEL + by * 32 + tx] = (_Float16)tile[tx][ty + i];
}

// ---------------------------------------------------------------------------
// ts[row] = clamp(tanh(x[row,:].Wo + bo))  (one wave per row, wave32 reduce)
// Clamp 1 ulp inside +-1 so the epilogue identity (tj-ti)/(1-tj*ti) is safe.
// ---------------------------------------------------------------------------
__global__ __launch_bounds__(256) void score_kernel(const float* __restrict__ x,
                                                    const float* __restrict__ Wo,
                                                    const float* __restrict__ bo,
                                                    float* __restrict__ ts) {
    const int lane = threadIdx.x & 31;
    const int wave = threadIdx.x >> 5;
    const int row  = blockIdx.x * 8 + wave;
    const float* xr = x + (size_t)row * D_MODEL;
    float acc = 0.f;
    for (int d = lane; d < D_MODEL; d += 32) acc += xr[d] * Wo[d];
    #pragma unroll
    for (int off = 16; off > 0; off >>= 1) acc += __shfl_xor(acc, off, 32);
    if (lane == 0) {
        float t = tanhf(acc + bo[0]);
        ts[row] = fminf(fmaxf(t, -0.99999988f), 0.99999988f);
    }
}

// ---------------------------------------------------------------------------
// WMMA operand builders per documented CDNA5 wave32 16-bit layouts.
// A (16x32 MxK): lane<16: M=lane, elems 0..7=K[0..7], 8..15=K[16..23]
//                lane>=16: M=lane-16, elems 0..7=K[8..15], 8..15=K[24..31]
// B (KxN) as kernels load it (A alike): lane<16: N=lane, elems=K[0..15]; lane>=16: N=lane-16, K[16..31]
// C/D: vgpr v, lane l -> row v+8*(l>>4), col l&15
// ---------------------------------------------------------------------------
__device__ inline v16h load_a_tile(const _Float16* p /* row m, + k0 + 8*half */) {
    v8h lo = *(const v8h*)(p);
    v8h hi = *(const v8h*)(p + 16);
    v16h a;
    #pragma unroll
    for (int i = 0; i < 8; ++i) { a[i] = lo[i]; a[8 + i] = hi[i]; }
    return a;
}

// ---------------------------------------------------------------------------
// Pass 1: Q = xh @ Wq + bq ; K = xh @ Wk + bk  (f16 in, f32 acc, f16 out)
// Register-blocked: each wave -> 64 rows x 32 cols (4x2 tiles, 8 WMMA/k-step).
// Block 256 thr = 8 waves in 2x4 -> 128 rows x 128 cols per block.
// grid = (ROWS/128, D_MODEL/128, 2)
// ---------------------------------------------------------------------------
__global__ __launch_bounds__(256) void proj_kernel(const _Float16* __restrict__ xh,
                                                   const _Float16* __restrict__ Wqt,
                                                   const _Float16* __restrict__ Wkt,
                                                   const float* __restrict__ bq,
                                                   const float* __restrict__ bk,
                                                   _Float16* __restrict__ Qh,
                                                   _Float16* __restrict__ Kh) {
    const int lane = threadIdx.x & 31;
    const int wave = threadIdx.x >> 5;
    const int wi = wave >> 2;            // 0..1
    const int wj = wave & 3;             // 0..3
    const int row0 = blockIdx.x * 128 + wi * 64;
    const int col0 = blockIdx.y * 128 + wj * 32;
    const int which = blockIdx.z;
    const _Float16* Wt  = which ? Wkt : Wqt;
    const float*    bia = which ? bk  : bq;
    _Float16*       dst = which ? Kh  : Qh;

    const int n    = lane & 15;
    const int half = lane >> 4;

    v8f acc[4][2] = {};
    for (int k0 = 0; k0 < D_MODEL; k0 += 32) {
        v16h a[4], bm[2];
        #pragma unroll
        for (int it = 0; it < 4; ++it)
            a[it] = *(const v16h*)(xh + (size_t)(row0 + 16 * it + n) * D_MODEL + k0 + 16 * half);
        #pragma unroll
        for (int jt = 0; jt < 2; ++jt)
            bm[jt] = *(const v16h*)(Wt + (size_t)(col0 + 16 * jt + n) * D_MODEL + k0 + 16 * half);
        #pragma unroll
        for (int it = 0; it < 4; ++it)
            #pragma unroll
            for (int jt = 0; jt < 2; ++jt)
                acc[it][jt] = __builtin_amdgcn_wmma_f32_16x16x32_f16(
                    false, a[it], false, bm[jt], (short)0, acc[it][jt], false, false);
    }
    #pragma unroll
    for (int jt = 0; jt < 2; ++jt) {
        const float bb = bia[col0 + 16 * jt + n];
        #pragma unroll
        for (int it = 0; it < 4; ++it)
            #pragma unroll
            for (int v = 0; v < 8; ++v) {
                int r = row0 + 16 * it + v + 8 * half;
                dst[(size_t)r * D_MODEL + col0 + 16 * jt + n] = (_Float16)(acc[it][jt][v] + bb);
            }
    }
}

// ---------------------------------------------------------------------------
// Pass 2: out[b,i,j] = (Q_i . K_j) * d^-0.5 + 5*(tj-ti)/(1-tj*ti)
// Same 4x2 register blocking; block = 8 waves -> 128x128 output tile.
// grid = (SEQ/128, SEQ/128, BATCH)
// ---------------------------------------------------------------------------
__global__ __launch_bounds__(256) void logits_kernel(const _Float16* __restrict__ Qh,
                                                     const _Float16* __restrict__ Kh,
                                                     const float* __restrict__ ts,
                                                     float* __restrict__ out) {
    const int lane = threadIdx.x & 31;
    const int wave = threadIdx.x >> 5;
    const int wi = wave >> 2;            // 0..1
    const int wj = wave & 3;             // 0..3
    const int b  = blockIdx.z;
    const int i0 = blockIdx.x * 128 + wi * 64;
    const int j0 = blockIdx.y * 128 + wj * 32;
    const int n    = lane & 15;
    const int half = lane >> 4;

    const _Float16* Qb = Qh + (size_t)b * SEQ * D_MODEL;
    const _Float16* Kb = Kh + (size_t)b * SEQ * D_MODEL;

    v8f acc[4][2] = {};
    for (int k0 = 0; k0 < D_MODEL; k0 += 32) {
        v16h a[4], bm[2];
        #pragma unroll
        for (int it = 0; it < 4; ++it)
            a[it] = *(const v16h*)(Qb + (size_t)(i0 + 16 * it + n) * D_MODEL + k0 + 16 * half);
        #pragma unroll
        for (int jt = 0; jt < 2; ++jt)
            bm[jt] = *(const v16h*)(Kb + (size_t)(j0 + 16 * jt + n) * D_MODEL + k0 + 16 * half);
        #pragma unroll
        for (int it = 0; it < 4; ++it)
            #pragma unroll
            for (int jt = 0; jt < 2; ++jt)
                acc[it][jt] = __builtin_amdgcn_wmma_f32_16x16x32_f16(
                    false, a[it], false, bm[jt], (short)0, acc[it][jt], false, false);
    }

    const float scale = 0.03125f;        // 1024^-0.5
    const float* tsb = ts + b * SEQ;
    float* outb = out + (size_t)b * SEQ * SEQ;
    #pragma unroll
    for (int it = 0; it < 4; ++it) {
        float ti[8];
        #pragma unroll
        for (int v = 0; v < 8; ++v) ti[v] = tsb[i0 + 16 * it + v + 8 * half];
        #pragma unroll
        for (int jt = 0; jt < 2; ++jt) {
            const int j = j0 + 16 * jt + n;
            const float tj = tsb[j];
            #pragma unroll
            for (int v = 0; v < 8; ++v) {
                const int i = i0 + 16 * it + v + 8 * half;
                // tanh(sj - si) = (tj - ti) / (1 - tj*ti)
                const float bias = 5.0f * (tj - ti[v]) / (1.0f - tj * ti[v]);
                outb[(size_t)i * SEQ + j] = acc[it][jt][v] * scale + bias;
            }
        }
    }
}

// ---------------------------------------------------------------------------
extern "C" void kernel_launch(void* const* d_in, const int* in_sizes, int n_in,
                              void* d_out, int out_size, void* d_ws, size_t ws_size,
                              hipStream_t stream) {
    const float* x  = (const float*)d_in[0];
    const float* Wq = (const float*)d_in[1];
    const float* bq = (const float*)d_in[2];
    const float* Wk = (const float*)d_in[3];
    const float* bk = (const float*)d_in[4];
    const float* Wo = (const float*)d_in[5];
    const float* bo = (const float*)d_in[6];
    float* out = (float*)d_out;

    // Workspace: xh, Qh, Kh (16 MB each f16), Wqt, Wkt (2 MB each f16), ts (32 KB f32)
    _Float16* xh  = (_Float16*)d_ws;
    _Float16* Qh  = xh  + (size_t)ROWS * D_MODEL;
    _Float16* Kh  = Qh  + (size_t)ROWS * D_MODEL;
    _Float16* Wqt = Kh  + (size_t)ROWS * D_MODEL;
    _Float16* Wkt = Wqt + (size_t)D_MODEL * D_MODEL;
    float*    ts  = (float*)(Wkt + (size_t)D_MODEL * D_MODEL);

    const int nx = ROWS * D_MODEL;                       // 8,388,608
    cvt_f16_kernel<<<nx / 256, 256, 0, stream>>>(x, xh, nx);
    dim3 gt(32, 32);
    transpose_f16_kernel<<<gt, 256, 0, stream>>>(Wq, Wqt);
    transpose_f16_kernel<<<gt, 256, 0, stream>>>(Wk, Wkt);
    score_kernel<<<ROWS / 8, 256, 0, stream>>>(x, Wo, bo, ts);

    dim3 g1(ROWS / 128, D_MODEL / 128, 2);               // (64, 8, 2)
    proj_kernel<<<g1, 256, 0, stream>>>(xh, Wqt, Wkt, bq, bk, Qh, Kh);

    dim3 g2(SEQ / 128, SEQ / 128, BATCH);                // (16, 16, 4)
    logits_kernel<<<g2, 256, 0, stream>>>(Qh, Kh, ts, out);
}
